// DualOrthogonalAttention_51762945851653
// MI455X (gfx1250) — compile-verified, matrix operand order checked by reading, not run on hardware
//
#include <hip/hip_runtime.h>
#include <hip/hip_bf16.h>
#include <math.h>

#define DIM     65
#define NHEADS  10
#define NODE    62
#define BSZ     1024

typedef __attribute__((ext_vector_type(16))) _Float16 v16h;
typedef __attribute__((ext_vector_type(8)))  _Float16 v8h;
typedef __attribute__((ext_vector_type(8)))  float    v8f;

// LDS strides (in halves / floats), padded for bank-conflict avoidance + 16B alignment
#define LDA   104   // 64x96 row-major f16 tiles (K padded to 96); 104*2B=208B, conflict-free rows
#define LDT   72    // 80x64 transposed f16 tiles (K padded to 64)
#define LDSS  68    // 64x64 f32 score tile
#define LDAT  72    // 64x64 f16 attention tiles

// Load a 16x32 f16 fragment from row-major LDS storage S[row][k].
// A-operand: rows are M.  For B = Y^T (computing X @ Y^T) the same loader works on Y
// with row0 = n-tile base (B layout assumed transpose-symmetric to documented A layout).
// Lane L: row = row0 + (L&15); chunks at k0 + 8*(L>>4) and +16 (CDNA5 ISA 7.12.2).
__device__ __forceinline__ v16h ld_frag(const _Float16* p, int ld, int row0, int k0) {
  const int lane = threadIdx.x & 31;
  const _Float16* base = p + (row0 + (lane & 15)) * ld + k0 + ((lane >> 4) << 3);
  const v8h lo = *(const v8h*)(base);
  const v8h hi = *(const v8h*)(base + 16);
  v16h r;
#pragma unroll
  for (int i = 0; i < 8; ++i) { r[i] = lo[i]; r[i + 8] = hi[i]; }
  return r;
}

__device__ __forceinline__ v8f wmma16(v16h a, v16h b, v8f c) {
  return __builtin_amdgcn_wmma_f32_16x16x32_f16(false, a, false, b, (short)0, c, false, false);
}

// ---------------- gate kernel: g[b] = sigmoid(cat(meanFn, meanFr) @ Wg^T + bg) ----------------
__global__ __launch_bounds__(160)
void gate_kernel(const float* __restrict__ Fn, const float* __restrict__ Fr,
                 const float* __restrict__ Wg, const float* __restrict__ bg,
                 float* __restrict__ g_ws, float* __restrict__ loss_slot) {
  const int b = blockIdx.x;
  const int t = threadIdx.x;
  __shared__ float cat[2 * DIM];
  if (b == 0 && t == 0) *loss_slot = 0.f;          // re-zeroed every launch (graph-safe)
  if (t < 2 * DIM) {
    const float* src = (t < DIM) ? (Fn + (size_t)b * NODE * DIM + t)
                                 : (Fr + (size_t)b * NODE * DIM + (t - DIM));
    float s = 0.f;
    for (int r = 0; r < NODE; ++r) s += src[r * DIM];
    cat[t] = s * (1.0f / NODE);
  }
  __syncthreads();
  if (t < DIM) {
    float acc = bg[t];
    const float* wrow = Wg + (size_t)t * 2 * DIM;
    for (int k = 0; k < 2 * DIM; ++k) acc += cat[k] * wrow[k];
    g_ws[(size_t)b * DIM + t] = 1.f / (1.f + __expf(-acc));
  }
}

// ---------------- main kernel: one block per (batch, head) ----------------
__global__ __launch_bounds__(256)
void dual_attn_kernel(const float* __restrict__ Fn, const float* __restrict__ Fr,
                      const float* __restrict__ Wq1, const float* __restrict__ bq1,
                      const float* __restrict__ Wk1, const float* __restrict__ bk1,
                      const float* __restrict__ Wq2, const float* __restrict__ bq2,
                      const float* __restrict__ Wk2, const float* __restrict__ bk2,
                      const float* __restrict__ g_ws,
                      float* __restrict__ out, float* __restrict__ loss_slot) {
  const int bh = blockIdx.x;
  const int b  = bh / NHEADS;
  const int h  = bh - b * NHEADS;
  const int tid  = threadIdx.x;
  const int wave = tid >> 5;
  const int lane = tid & 31;

  __shared__ __align__(16) _Float16 sFn [64 * LDA];   // Fn[b] f16, K-padded to 96
  __shared__ __align__(16) _Float16 sFr [64 * LDA];
  __shared__ __align__(16) _Float16 sFnT[80 * LDT];   // Fn[b]^T (d-major), K-padded to 64
  __shared__ __align__(16) _Float16 sFrT[80 * LDT];
  __shared__ __align__(16) _Float16 sW  [80 * LDA];   // current W_h slice (j x k), zero-padded
  __shared__ __align__(16) _Float16 sQ  [64 * LDA];   // projection results
  __shared__ __align__(16) _Float16 sK  [64 * LDA];
  __shared__ __align__(16) float    sS  [64 * LDSS];  // raw scores (f32)
  __shared__ __align__(16) _Float16 sA1 [64 * LDAT];  // softmaxed attentions (f16)
  __shared__ __align__(16) _Float16 sA2 [64 * LDAT];
  __shared__ float sBias[80];
  __shared__ float sGv[80];
  __shared__ float sRed;

  // ---- stage inputs, zero scratch ----
  if (tid == 0) sRed = 0.f;
  for (int i = tid; i < 80; i += 256) sGv[i] = (i < DIM) ? g_ws[(size_t)b * DIM + i] : 0.f;
  const float* FnB = Fn + (size_t)b * NODE * DIM;
  const float* FrB = Fr + (size_t)b * NODE * DIM;
  for (int i = tid; i < 64 * 96; i += 256) {
    int r = i / 96, k = i - r * 96;
    bool ok = (r < NODE) & (k < DIM);
    sFn[r * LDA + k] = (_Float16)(ok ? FnB[r * DIM + k] : 0.f);
    sFr[r * LDA + k] = (_Float16)(ok ? FrB[r * DIM + k] : 0.f);
  }
  for (int i = tid; i < 80 * 64; i += 256) {
    int d = i / 64, r = i - d * 64;
    bool ok = (d < DIM) & (r < NODE);
    sFnT[d * LDT + r] = (_Float16)(ok ? FnB[r * DIM + d] : 0.f);
    sFrT[d * LDT + r] = (_Float16)(ok ? FrB[r * DIM + d] : 0.f);
  }
  for (int i = tid; i < 64 * LDA; i += 256) { sQ[i] = (_Float16)0.f; sK[i] = (_Float16)0.f; }
  __syncthreads();

  // ---- stage a per-head weight slice: sW[j][k] = W[(h*65+j)*65+k], zero padded ----
  auto stageW = [&](const float* W, const float* bv) {
    const float* Wh = W + (size_t)h * DIM * DIM;
    for (int i = tid; i < 80 * 96; i += 256) {
      int j = i / 96, k = i - j * 96;
      sW[j * LDA + k] = (_Float16)(((j < DIM) & (k < DIM)) ? Wh[j * DIM + k] : 0.f);
    }
    for (int i = tid; i < 80; i += 256) sBias[i] = (i < DIM) ? bv[h * DIM + i] : 0.f;
  };

  // ---- projection: dst(64x80) = src(64x96) @ sW^T + bias ----
  auto proj = [&](const _Float16* src, _Float16* dst) {
    for (int t = wave; t < 20; t += 8) {
      int tm = t / 5, tn = t - tm * 5;
      v8f acc = {};
#pragma unroll
      for (int kk = 0; kk < 3; ++kk) {
        v16h a  = ld_frag(src, LDA, tm * 16, kk * 32);
        v16h bf = ld_frag(sW,  LDA, tn * 16, kk * 32);   // B = W^T
        acc = wmma16(a, bf, acc);
      }
      int n = tn * 16 + (lane & 15);
      float bias = sBias[n];
      int mbase = tm * 16 + ((lane >> 4) << 3);
#pragma unroll
      for (int r = 0; r < 8; ++r)
        dst[(mbase + r) * LDA + n] = (_Float16)(acc[r] + bias);
    }
  };

  // ---- scores: sS(64x64) = sQ @ sK^T (raw, f32) ----
  auto scores = [&]() {
    for (int t = wave; t < 16; t += 8) {
      int tm = t >> 2, tn = t & 3;
      v8f acc = {};
#pragma unroll
      for (int kk = 0; kk < 3; ++kk) {
        v16h a  = ld_frag(sQ, LDA, tm * 16, kk * 32);
        v16h bf = ld_frag(sK, LDA, tn * 16, kk * 32);
        acc = wmma16(a, bf, acc);
      }
      int n = tn * 16 + (lane & 15);
      int mbase = tm * 16 + ((lane >> 4) << 3);
#pragma unroll
      for (int r = 0; r < 8; ++r) sS[(mbase + r) * LDSS + n] = acc[r];
    }
  };

  // ---- masked row softmax (cols < 62), write f16 with zeroed padding ----
  auto softmaxTo = [&](_Float16* dA) {
    const float scale = 0.124034734589208f;  // 1/sqrt(65)
    if (tid < NODE) {
      int row = tid;
      float m = -1e30f;
      for (int c = 0; c < NODE; ++c) m = fmaxf(m, sS[row * LDSS + c] * scale);
      float sum = 0.f;
      for (int c = 0; c < NODE; ++c) sum += __expf(sS[row * LDSS + c] * scale - m);
      float inv = 1.f / sum;
      for (int c = 0; c < NODE; ++c)
        dA[row * LDAT + c] = (_Float16)(__expf(sS[row * LDSS + c] * scale - m) * inv);
      dA[row * LDAT + 62] = (_Float16)0.f;
      dA[row * LDAT + 63] = (_Float16)0.f;
    } else if (tid < 64) {
      for (int c = 0; c < 64; ++c) dA[tid * LDAT + c] = (_Float16)0.f;
    }
  };

  // ================= pipeline =================
  stageW(Wq1, bq1);  __syncthreads();
  proj(sFn, sQ);     __syncthreads();
  stageW(Wk1, bk1);  __syncthreads();
  proj(sFr, sK);     __syncthreads();
  scores();          __syncthreads();
  softmaxTo(sA1);    __syncthreads();

  stageW(Wq2, bq2);  __syncthreads();
  proj(sFr, sQ);     __syncthreads();
  stageW(Wk2, bk2);  __syncthreads();
  proj(sFn, sK);     __syncthreads();
  scores();          __syncthreads();
  softmaxTo(sA2);    __syncthreads();

  // ---- C = A1 @ A2^T ; accumulate ||C - I||_F^2 ----
  {
    float local = 0.f;
    for (int t = wave; t < 16; t += 8) {
      int tm = t >> 2, tn = t & 3;
      v8f acc = {};
#pragma unroll
      for (int kk = 0; kk < 2; ++kk) {
        v16h a  = ld_frag(sA1, LDAT, tm * 16, kk * 32);
        v16h bf = ld_frag(sA2, LDAT, tn * 16, kk * 32);
        acc = wmma16(a, bf, acc);
      }
      int n = tn * 16 + (lane & 15);
      int mbase = tm * 16 + ((lane >> 4) << 3);
#pragma unroll
      for (int r = 0; r < 8; ++r) {
        int m = mbase + r;
        if ((m < NODE) & (n < NODE)) {
          float d = acc[r] - ((m == n) ? 1.f : 0.f);
          local += d * d;
        }
      }
    }
    atomicAdd(&sRed, local);   // ds_add_f32
  }

  // ---- gated outputs: out_n = g * (A1 @ Fr), out_r = (1-g) * (A2 @ Fn) ----
  auto outStage = [&](const _Float16* A, const _Float16* FT, int rowOff, bool useG) {
    for (int t = wave; t < 20; t += 8) {
      int tm = t / 5, tn = t - tm * 5;
      v8f acc = {};
#pragma unroll
      for (int kk = 0; kk < 2; ++kk) {
        v16h a  = ld_frag(A,  LDAT, tm * 16, kk * 32);
        v16h bf = ld_frag(FT, LDT,  tn * 16, kk * 32);  // B = F (KxN) via transposed copy
        acc = wmma16(a, bf, acc);
      }
      int n = tn * 16 + (lane & 15);
      if (n < DIM) {
        float gv = useG ? sGv[n] : (1.f - sGv[n]);
        int mbase = tm * 16 + ((lane >> 4) << 3);
#pragma unroll
        for (int r = 0; r < 8; ++r) {
          int m = mbase + r;
          if (m < NODE)
            out[((size_t)bh * 124 + rowOff + m) * DIM + n] = gv * acc[r];
        }
      }
    }
  };
  outStage(sA1, sFrT, 0,  true);
  outStage(sA2, sFnT, 62, false);

  __syncthreads();
  if (tid == 0)
    atomicAdd(loss_slot, sqrtf(sRed) * (1.f / (float)(BSZ * NHEADS)));
}

// ---------------- launch ----------------
extern "C" void kernel_launch(void* const* d_in, const int* in_sizes, int n_in,
                              void* d_out, int out_size, void* d_ws, size_t ws_size,
                              hipStream_t stream) {
  const float* Fn  = (const float*)d_in[0];
  const float* Fr  = (const float*)d_in[1];
  const float* Wq1 = (const float*)d_in[2];
  const float* bq1 = (const float*)d_in[3];
  const float* Wk1 = (const float*)d_in[4];
  const float* bk1 = (const float*)d_in[5];
  const float* Wq2 = (const float*)d_in[6];
  const float* bq2 = (const float*)d_in[7];
  const float* Wk2 = (const float*)d_in[8];
  const float* bk2 = (const float*)d_in[9];
  const float* Wg  = (const float*)d_in[10];
  const float* bg  = (const float*)d_in[11];

  float* out = (float*)d_out;
  float* loss_slot = out + (size_t)BSZ * NHEADS * 124 * DIM;  // scalar after fused
  float* g_ws = (float*)d_ws;                                  // BSZ*DIM floats

  gate_kernel<<<BSZ, 160, 0, stream>>>(Fn, Fr, Wg, bg, g_ws, loss_slot);
  dual_attn_kernel<<<BSZ * NHEADS, 256, 0, stream>>>(
      Fn, Fr, Wq1, bq1, Wk1, bk1, Wq2, bq2, Wk2, bk2, g_ws, out, loss_slot);
}
